// MLAv3_59940563583326
// MI455X (gfx1250) — compile-verified, matrix operand order checked by reading, not run on hardware
//
#include <hip/hip_runtime.h>

// MLA forward for MI455X (gfx1250, wave32).
// - All GEMMs + attention on v_wmma_f32_16x16x32_bf16 (f32 accumulate).
// - Attention K/V tiles staged via the Tensor Data Mover
//   (tensor_load_to_lds + s_wait_tensorcnt), double-buffered in LDS.

typedef __attribute__((ext_vector_type(16))) __bf16    v16bf;
typedef __attribute__((ext_vector_type(8)))  float     v8f;
typedef __attribute__((ext_vector_type(4)))  unsigned  v4u;
typedef __attribute__((ext_vector_type(8)))  int       v8i;
typedef __attribute__((ext_vector_type(4)))  int       v4i;

#define NUM_HEAD   16
#define EMBD_DIM   2048
#define LATENT_DIM 512
#define ROPE_DIM   64
#define HEAD_DIM   128
#define QK_DIM     192           // HEAD_DIM + ROPE_DIM
#define ATT_SCALE  0.07216878364870323f  // (192)^-0.5

#if __has_builtin(__builtin_amdgcn_tensor_load_to_lds)
#define MLA_TDM 1
#else
#define MLA_TDM 0
#endif

static __device__ __forceinline__ __bf16 f2bf(float f) {
  unsigned u = __builtin_bit_cast(unsigned, f);
  unsigned r = (u + 0x7FFFu + ((u >> 16) & 1u)) >> 16;   // round-to-nearest-even
  unsigned short s = (unsigned short)r;
  return __builtin_bit_cast(__bf16, s);
}
static __device__ __forceinline__ float bf2f(__bf16 h) {
  unsigned short s = __builtin_bit_cast(unsigned short, h);
  unsigned u = ((unsigned)s) << 16;
  return __builtin_bit_cast(float, u);
}

// A-fragment (and mirrored B^T-fragment) K index for vgpr-pair v, half-wave lh.
static __device__ __forceinline__ int frag_kk(int v, int lh) {
  return ((v >> 2) << 4) + (lh << 3) + ((v & 3) << 1);
}

#if MLA_TDM
// 2D TDM load: tile of `rows` x `cols` elements (2-byte bf16), row stride
// `stride` elements, from global `gaddr` into LDS byte offset `lds_addr`.
// D# layout per CDNA5 ISA ch.8 (group0: control/lds/global; group1: dims).
static __device__ __forceinline__ void tdm_load_2d_bf16(
    unsigned lds_addr, const void* gaddr,
    unsigned rows, unsigned cols, unsigned stride)
{
  unsigned long long ga = (unsigned long long)(uintptr_t)gaddr;
  v4u g0;
  g0[0] = 1u;                                        // count=1 (valid), user mode
  g0[1] = lds_addr;                                  // LDS byte address
  g0[2] = (unsigned)(ga & 0xFFFFFFFFu);              // global_addr[31:0]
  g0[3] = (unsigned)((ga >> 32) & 0x01FFFFFFu)       // global_addr[56:32]
          | (2u << 30);                              // type=2 ("image")
  v8i g1;
  g1[0] = (int)(1u << 16);                           // data_size=1 (2B), wg_mask=0
  g1[1] = (int)((cols & 0xFFFFu) << 16);             // tensor_dim0[15:0]
  g1[2] = (int)((rows & 0xFFFFu) << 16);             // dim0 hi=0 | tensor_dim1 lo
  g1[3] = (int)((cols & 0xFFFFu) << 16);             // dim1 hi=0 | tile_dim0
  g1[4] = (int)(rows & 0xFFFFu);                     // tile_dim1 | tile_dim2=0
  g1[5] = (int)stride;                               // tensor_dim0_stride[31:0]
  g1[6] = 0;
  g1[7] = 0;
  v4i gz = {0, 0, 0, 0};
#if __clang_major__ >= 23
  v8i gz8 = {0, 0, 0, 0, 0, 0, 0, 0};
  __builtin_amdgcn_tensor_load_to_lds(g0, g1, gz, gz, gz8, 0);
#else
  __builtin_amdgcn_tensor_load_to_lds(g0, g1, gz, gz, 0);
#endif
}
#endif

// ---------------------------------------------------------------------------
// Generic tiled GEMM: C[M,N] = A[M,K] (f32) * B[K,N] (f32), bf16 WMMA compute.
// Block tile 128x64, 8 waves, each wave a 32x32 subtile (2x2 WMMA tiles).
// MODE 0: write f32 contiguous. MODE 1: write bf16 scattered by head.
// ---------------------------------------------------------------------------
template<int MODE>
__global__ __launch_bounds__(256) void mla_gemm(
    const float* __restrict__ A, const float* __restrict__ B,
    float* __restrict__ Cf, __bf16* __restrict__ Cbf,
    int M, int N, int K, int S,
    int head_shift, int ostride, int ooffset)
{
  __shared__ __bf16 As[128][40];   // 128x32 tile, padded
  __shared__ __bf16 Bs[32][72];    // 32x64 tile, padded

  const int tid  = threadIdx.x;
  const int wid  = tid >> 5;
  const int lane = tid & 31;
  const int lm   = lane & 15;
  const int lh   = lane >> 4;
  const int wm   = wid >> 1;       // 0..3 (M direction)
  const int wn   = wid & 1;        // 0..1 (N direction)
  const int m0   = blockIdx.y * 128;
  const int n0   = blockIdx.x * 64;

  v8f acc[2][2];
#pragma unroll
  for (int i = 0; i < 2; ++i)
#pragma unroll
    for (int j = 0; j < 2; ++j)
#pragma unroll
      for (int v = 0; v < 8; ++v) acc[i][j][v] = 0.f;

  for (int k0 = 0; k0 < K; k0 += 32) {
    // Stage A tile as float4 loads, converting f32 -> bf16 into LDS.
#pragma unroll
    for (int i = tid; i < 128 * 8; i += 256) {
      const int r = i >> 3, c4 = (i & 7) << 2;
      const float4 f = *(const float4*)&A[(size_t)(m0 + r) * K + (k0 + c4)];
      As[r][c4 + 0] = f2bf(f.x);
      As[r][c4 + 1] = f2bf(f.y);
      As[r][c4 + 2] = f2bf(f.z);
      As[r][c4 + 3] = f2bf(f.w);
    }
#pragma unroll
    for (int i = tid; i < 32 * 16; i += 256) {
      const int r = i >> 4, c4 = (i & 15) << 2;
      const float4 f = *(const float4*)&B[(size_t)(k0 + r) * N + (n0 + c4)];
      Bs[r][c4 + 0] = f2bf(f.x);
      Bs[r][c4 + 1] = f2bf(f.y);
      Bs[r][c4 + 2] = f2bf(f.z);
      Bs[r][c4 + 3] = f2bf(f.w);
    }
    if (k0 + 32 < K) {
      __builtin_prefetch(&A[(size_t)(m0 + (tid >> 1)) * K + (k0 + 32)], 0, 1);
      __builtin_prefetch(&B[(size_t)(k0 + 32 + (tid & 31)) * N + n0], 0, 1);
    }
    __syncthreads();

    v16bf af[2], bfm[2];
#pragma unroll
    for (int t = 0; t < 2; ++t) {
      const int mrow = wm * 32 + t * 16 + lm;
      const int ncol = wn * 32 + t * 16 + lm;
#pragma unroll
      for (int v = 0; v < 8; ++v) {
        const int kk = frag_kk(v, lh);
        af[t][2 * v]      = As[mrow][kk];
        af[t][2 * v + 1]  = As[mrow][kk + 1];
        bfm[t][2 * v]     = Bs[kk][ncol];
        bfm[t][2 * v + 1] = Bs[kk + 1][ncol];
      }
    }
#pragma unroll
    for (int i = 0; i < 2; ++i)
#pragma unroll
      for (int j = 0; j < 2; ++j)
        acc[i][j] = __builtin_amdgcn_wmma_f32_16x16x32_bf16(
            false, af[i], false, bfm[j], (short)0, acc[i][j], false, false);
    __syncthreads();
  }

  // Epilogue. C/D layout: element v -> row (v + 8*lh) in tile, col = lm.
#pragma unroll
  for (int i = 0; i < 2; ++i)
#pragma unroll
    for (int j = 0; j < 2; ++j)
#pragma unroll
      for (int v = 0; v < 8; ++v) {
        const int row = m0 + wm * 32 + i * 16 + v + (lh << 3);
        const int col = n0 + wn * 32 + j * 16 + lm;
        const float val = acc[i][j][v];
        if (MODE == 0) {
          Cf[(size_t)row * N + col] = val;
        } else {
          const int bidx = row / S;
          const int s    = row - bidx * S;
          const int h    = col >> head_shift;
          const int d    = col & ((1 << head_shift) - 1);
          Cbf[((size_t)(bidx * NUM_HEAD + h) * S + s) * (size_t)ostride
              + ooffset + d] = f2bf(val);
        }
      }
}

// ---------------------------------------------------------------------------
// RoPE on q_r region of Q[B,H,S,192] (cols 128..191), interleaved pairs.
// ---------------------------------------------------------------------------
__global__ __launch_bounds__(256) void rope_q_kernel(__bf16* __restrict__ Qg,
                                                     int S, int total)
{
  const int idx = blockIdx.x * 256 + threadIdx.x;
  if (idx >= total) return;
  const int j    = idx & 31;          // freq index, 0..31
  const int rest = idx >> 5;
  const int s    = rest % S;
  const int bh   = rest / S;
  const size_t base = ((size_t)bh * S + s) * QK_DIM + HEAD_DIM + 2 * j;
  const float xe = bf2f(Qg[base]);
  const float xo = bf2f(Qg[base + 1]);
  const float inv = __powf(10000.f, -(float)(2 * j) / (float)ROPE_DIM);
  const float ang = (float)s * inv;
  const float c = __cosf(ang), sn = __sinf(ang);
  Qg[base]     = f2bf(xe * c - xo * sn);
  Qg[base + 1] = f2bf(xe * sn + xo * c);
}

// ---------------------------------------------------------------------------
// RoPE on k_r (f32 [B*S,64]) + broadcast into all heads of K[B,H,S,192].
// ---------------------------------------------------------------------------
__global__ __launch_bounds__(256) void rope_k_kernel(const float* __restrict__ kr,
                                                     __bf16* __restrict__ Kg,
                                                     int S, int total)
{
  const int idx = blockIdx.x * 256 + threadIdx.x;
  if (idx >= total) return;
  const int j   = idx & 31;
  const int row = idx >> 5;           // b*S + s
  const int s   = row % S;
  const int b   = row / S;
  const float xe = kr[(size_t)row * ROPE_DIM + 2 * j];
  const float xo = kr[(size_t)row * ROPE_DIM + 2 * j + 1];
  const float inv = __powf(10000.f, -(float)(2 * j) / (float)ROPE_DIM);
  const float ang = (float)s * inv;
  const float c = __cosf(ang), sn = __sinf(ang);
  const __bf16 he = f2bf(xe * c - xo * sn);
  const __bf16 ho = f2bf(xe * sn + xo * c);
#pragma unroll
  for (int h = 0; h < NUM_HEAD; ++h) {
    const size_t base = ((size_t)(b * NUM_HEAD + h) * S + s) * QK_DIM
                        + HEAD_DIM + 2 * j;
    Kg[base]     = he;
    Kg[base + 1] = ho;
  }
}

// ---------------------------------------------------------------------------
// Flash attention, causal. One block = 128 query rows of one (b,h).
// 8 waves x 16 rows each. K/V tiles (32 keys) double-buffered in LDS and
// filled by the Tensor Data Mover (wave 0 issues, s_wait_tensorcnt + barrier
// publishes). QK^T: 6 WMMAs per 16x16 score tile; PV: 8 WMMAs per key block.
// ---------------------------------------------------------------------------
__global__ __launch_bounds__(256) void mla_attn(
    const __bf16* __restrict__ Qg, const __bf16* __restrict__ Kg,
    const __bf16* __restrict__ Vg, float* __restrict__ Og, int S)
{
  __shared__ __bf16 Ks[2][32][QK_DIM];     // 2 x 32 keys x 192
  __shared__ __bf16 Vs[2][32][HEAD_DIM];   // 2 x 32 keys x 128
  __shared__ __bf16 Ps[8][16][32];         // per-wave P tile 16x32

  const int tid  = threadIdx.x;
  const int wid  = tid >> 5;
  const int lane = tid & 31;
  const int lm   = lane & 15;
  const int lh   = lane >> 4;
  const int bh   = blockIdx.y;
  const int b    = bh >> 4;
  const int h    = bh & 15;
  const int q0   = blockIdx.x * 128;
  const size_t kvbase = (size_t)bh * S;

  // Resident Q fragments: 6 d-tiles of 16x32 for this wave's 16 rows.
  const int qrow = q0 + wid * 16 + lm;
  const __bf16* qptr = Qg + (kvbase + qrow) * QK_DIM;
  v16bf qa[6];
#pragma unroll
  for (int t = 0; t < 6; ++t)
#pragma unroll
    for (int v = 0; v < 8; ++v) {
      const int kk = frag_kk(v, lh);
      qa[t][2 * v]     = qptr[t * 32 + kk];
      qa[t][2 * v + 1] = qptr[t * 32 + kk + 1];
    }

  v8f o[8];
#pragma unroll
  for (int t = 0; t < 8; ++t)
#pragma unroll
    for (int v = 0; v < 8; ++v) o[t][v] = 0.f;
  float mrun[8], lrun[8];
#pragma unroll
  for (int v = 0; v < 8; ++v) { mrun[v] = -__builtin_inff(); lrun[v] = 0.f; }

  const int nkb = (q0 + 127) / 32 + 1;     // causal: only key blocks <= row max

  // Prologue: fill buffer 0 with key block 0.
#if MLA_TDM
  if (wid == 0) {
    tdm_load_2d_bf16((unsigned)(uintptr_t)&Ks[0][0][0],
                     Kg + kvbase * QK_DIM, 32, QK_DIM, QK_DIM);
    tdm_load_2d_bf16((unsigned)(uintptr_t)&Vs[0][0][0],
                     Vg + kvbase * HEAD_DIM, 32, HEAD_DIM, HEAD_DIM);
    __builtin_amdgcn_s_wait_tensorcnt(0);
  }
#else
  for (int i = tid; i < 32 * QK_DIM / 8; i += 256) {
    const int r = i / (QK_DIM / 8), c = (i % (QK_DIM / 8)) * 8;
    *(uint4*)&Ks[0][r][c] = *(const uint4*)&Kg[(kvbase + r) * QK_DIM + c];
  }
  for (int i = tid; i < 32 * HEAD_DIM / 8; i += 256) {
    const int r = i >> 4, c = (i & 15) * 8;
    *(uint4*)&Vs[0][r][c] = *(const uint4*)&Vg[(kvbase + r) * HEAD_DIM + c];
  }
#endif
  __syncthreads();

  for (int kb = 0; kb < nkb; ++kb) {
    const int kbase = kb * 32;
    const int cur = kb & 1;
    const int nxt = cur ^ 1;

    // Kick off the async fill of the next key block while we compute.
    if (kb + 1 < nkb) {
#if MLA_TDM
      if (wid == 0) {
        tdm_load_2d_bf16((unsigned)(uintptr_t)&Ks[nxt][0][0],
                         Kg + (kvbase + kbase + 32) * QK_DIM, 32, QK_DIM, QK_DIM);
        tdm_load_2d_bf16((unsigned)(uintptr_t)&Vs[nxt][0][0],
                         Vg + (kvbase + kbase + 32) * HEAD_DIM, 32, HEAD_DIM, HEAD_DIM);
      }
#else
      for (int i = tid; i < 32 * QK_DIM / 8; i += 256) {
        const int r = i / (QK_DIM / 8), c = (i % (QK_DIM / 8)) * 8;
        *(uint4*)&Ks[nxt][r][c] =
            *(const uint4*)&Kg[(kvbase + kbase + 32 + r) * QK_DIM + c];
      }
      for (int i = tid; i < 32 * HEAD_DIM / 8; i += 256) {
        const int r = i >> 4, c = (i & 15) * 8;
        *(uint4*)&Vs[nxt][r][c] =
            *(const uint4*)&Vg[(kvbase + kbase + 32 + r) * HEAD_DIM + c];
      }
#endif
    }

    // Scores: 16 rows x 32 keys as two 16x16 C tiles.
    v8f sc[2];
#pragma unroll
    for (int j = 0; j < 2; ++j)
#pragma unroll
      for (int v = 0; v < 8; ++v) sc[j][v] = 0.f;
#pragma unroll
    for (int j = 0; j < 2; ++j) {
#pragma unroll
      for (int t = 0; t < 6; ++t) {
        v16bf kf;
#pragma unroll
        for (int v = 0; v < 8; ++v) {
          const int kk = frag_kk(v, lh);
          kf[2 * v]     = Ks[cur][j * 16 + lm][t * 32 + kk];
          kf[2 * v + 1] = Ks[cur][j * 16 + lm][t * 32 + kk + 1];
        }
        sc[j] = __builtin_amdgcn_wmma_f32_16x16x32_bf16(
            false, qa[t], false, kf, (short)0, sc[j], false, false);
      }
    }

    // Online softmax. Row of element v is (v + 8*lh); cols live across the
    // 16-lane half-group, so row reductions are shfl_xor over offsets 1..8.
#pragma unroll
    for (int v = 0; v < 8; ++v) {
      const int qr = q0 + wid * 16 + v + (lh << 3);
      float s0 = sc[0][v] * ATT_SCALE;
      float s1 = sc[1][v] * ATT_SCALE;
      if (kbase + lm > qr)      s0 = -__builtin_inff();
      if (kbase + 16 + lm > qr) s1 = -__builtin_inff();
      float rm = fmaxf(s0, s1);
#pragma unroll
      for (int d = 1; d < 16; d <<= 1) rm = fmaxf(rm, __shfl_xor(rm, d, 32));
      const float mnew  = fmaxf(mrun[v], rm);
      const float alpha = __expf(mrun[v] - mnew);
      const float p0 = __expf(s0 - mnew);
      const float p1 = __expf(s1 - mnew);
      float rs = p0 + p1;
#pragma unroll
      for (int d = 1; d < 16; d <<= 1) rs += __shfl_xor(rs, d, 32);
      lrun[v] = lrun[v] * alpha + rs;
      mrun[v] = mnew;
      Ps[wid][v + (lh << 3)][lm]      = f2bf(p0);
      Ps[wid][v + (lh << 3)][16 + lm] = f2bf(p1);
#pragma unroll
      for (int t = 0; t < 8; ++t) o[t][v] *= alpha;
    }
    __syncthreads();                        // P relayout visibility

    // P as A-fragment (16x32), V column tiles as B-fragments; 8 WMMAs.
    v16bf pa;
#pragma unroll
    for (int v = 0; v < 8; ++v) {
      const int kk = frag_kk(v, lh);
      pa[2 * v]     = Ps[wid][lm][kk];
      pa[2 * v + 1] = Ps[wid][lm][kk + 1];
    }
#pragma unroll
    for (int t = 0; t < 8; ++t) {
      v16bf vf;
#pragma unroll
      for (int v = 0; v < 8; ++v) {
        const int kk = frag_kk(v, lh);
        vf[2 * v]     = Vs[cur][kk][t * 16 + lm];
        vf[2 * v + 1] = Vs[cur][kk + 1][t * 16 + lm];
      }
      o[t] = __builtin_amdgcn_wmma_f32_16x16x32_bf16(
          false, pa, false, vf, (short)0, o[t], false, false);
    }

    // Publish next buffer: wave 0 drains TENSORcnt, everyone syncs.
#if MLA_TDM
    if (wid == 0) __builtin_amdgcn_s_wait_tensorcnt(0);
#endif
    __syncthreads();
  }

  // Normalize and write f32 output at [b, s, h*128 + col].
#pragma unroll
  for (int t = 0; t < 8; ++t)
#pragma unroll
    for (int v = 0; v < 8; ++v) {
      const int qr = q0 + wid * 16 + v + (lh << 3);
      const float val = o[t][v] / lrun[v];
      Og[((size_t)b * S + qr) * EMBD_DIM + h * HEAD_DIM + t * 16 + lm] = val;
    }
}

// ---------------------------------------------------------------------------
// Host launch
// ---------------------------------------------------------------------------
extern "C" void kernel_launch(void* const* d_in, const int* in_sizes, int n_in,
                              void* d_out, int out_size, void* d_ws, size_t ws_size,
                              hipStream_t stream)
{
  const float* x     = (const float*)d_in[0];
  const float* w_ckv = (const float*)d_in[1];
  const float* w_k   = (const float*)d_in[2];
  const float* w_v   = (const float*)d_in[3];
  const float* w_cq  = (const float*)d_in[4];
  const float* w_q   = (const float*)d_in[5];
  const float* w_qr  = (const float*)d_in[6];
  const float* w_kr  = (const float*)d_in[7];
  const float* w_o   = (const float*)d_in[8];
  float* out = (float*)d_out;

  const int M  = in_sizes[0] / EMBD_DIM;   // B*S = 4096
  const int Bn = 2;
  const int S  = M / Bn;                   // 2048

  // Workspace carve-up (256B aligned)
  char* ws = (char*)d_ws;
  size_t off = 0;
  auto carve = [&](size_t bytes) -> char* {
    char* p = ws + off;
    off += (bytes + 255) & ~(size_t)255;
    return p;
  };
  float*  c_kv   = (float*) carve((size_t)M * LATENT_DIM * sizeof(float));
  float*  c_q    = (float*) carve((size_t)M * LATENT_DIM * sizeof(float));
  float*  krtmp  = (float*) carve((size_t)M * ROPE_DIM * sizeof(float));
  __bf16* Qb     = (__bf16*)carve((size_t)Bn * NUM_HEAD * S * QK_DIM * 2);
  __bf16* Kb     = (__bf16*)carve((size_t)Bn * NUM_HEAD * S * QK_DIM * 2);
  __bf16* Vb     = (__bf16*)carve((size_t)Bn * NUM_HEAD * S * HEAD_DIM * 2);
  float*  attn_o = (float*) carve((size_t)M * EMBD_DIM * sizeof(float));
  (void)ws_size; (void)n_in; (void)out_size;

  const dim3 blk(256);

  // 1-2: latent projections (f32 out)
  mla_gemm<0><<<dim3(LATENT_DIM / 64, M / 128), blk, 0, stream>>>(
      x, w_ckv, c_kv, nullptr, M, LATENT_DIM, EMBD_DIM, S, 0, 0, 0);
  mla_gemm<0><<<dim3(LATENT_DIM / 64, M / 128), blk, 0, stream>>>(
      x, w_cq, c_q, nullptr, M, LATENT_DIM, EMBD_DIM, S, 0, 0, 0);
  // 3: k_r pre-rope (f32 out)
  mla_gemm<0><<<dim3(ROPE_DIM / 64, M / 128), blk, 0, stream>>>(
      x, w_kr, krtmp, nullptr, M, ROPE_DIM, EMBD_DIM, S, 0, 0, 0);
  // 4: k_c -> K[:, :, :, 0:128] bf16
  mla_gemm<1><<<dim3(EMBD_DIM / 64, M / 128), blk, 0, stream>>>(
      c_kv, w_k, nullptr, Kb, M, EMBD_DIM, LATENT_DIM, S, 7, QK_DIM, 0);
  // 5: v -> V bf16
  mla_gemm<1><<<dim3(EMBD_DIM / 64, M / 128), blk, 0, stream>>>(
      c_kv, w_v, nullptr, Vb, M, EMBD_DIM, LATENT_DIM, S, 7, HEAD_DIM, 0);
  // 6: q_c -> Q[:, :, :, 0:128] bf16
  mla_gemm<1><<<dim3(EMBD_DIM / 64, M / 128), blk, 0, stream>>>(
      c_q, w_q, nullptr, Qb, M, EMBD_DIM, LATENT_DIM, S, 7, QK_DIM, 0);
  // 7: q_r -> Q[:, :, :, 128:192] bf16
  mla_gemm<1><<<dim3((NUM_HEAD * ROPE_DIM) / 64, M / 128), blk, 0, stream>>>(
      c_q, w_qr, nullptr, Qb, M, NUM_HEAD * ROPE_DIM, LATENT_DIM, S,
      6, QK_DIM, HEAD_DIM);

  // 8: RoPE(q_r)
  {
    const long total = (long)Bn * NUM_HEAD * S * (ROPE_DIM / 2);
    rope_q_kernel<<<(unsigned)((total + 255) / 256), blk, 0, stream>>>(
        Qb, S, (int)total);
  }
  // 9: RoPE(k_r) + broadcast into K heads
  {
    const long total = (long)M * (ROPE_DIM / 2);
    rope_k_kernel<<<(unsigned)((total + 255) / 256), blk, 0, stream>>>(
        krtmp, Kb, S, (int)total);
  }

  // 10: causal flash attention -> attn_o f32 [B,S,2048]
  mla_attn<<<dim3(S / 128, Bn * NUM_HEAD), blk, 0, stream>>>(
      Qb, Kb, Vb, attn_o, S);

  // 11: output projection -> d_out f32
  mla_gemm<0><<<dim3(EMBD_DIM / 64, M / 128), blk, 0, stream>>>(
      attn_o, w_o, out, nullptr, M, EMBD_DIM, EMBD_DIM, S, 0, 0, 0);
}
